// LSTM_Cell_63505386439276
// MI455X (gfx1250) — matrix operand order repaired and checked by reading, not compiled, not run on hardware
//
#include <hip/hip_runtime.h>
#include <hip/hip_bf16.h>
#include <stdint.h>

// ---------------------------------------------------------------------------
// LSTM cell, B=4096, D_IN=2048, UNITS=2048, D_CAT=4096
//   z = concat(h, x) @ concat(Wf,Wi,Wo,Wc) + b      -> fused bf16 WMMA GEMM
//   gates + cell update fused into the GEMM epilogue
//   Staging: CDNA5 async global->LDS (ASYNCcnt), triple-buffered.
// ---------------------------------------------------------------------------

typedef __attribute__((ext_vector_type(16))) __bf16 v16bf;
typedef __attribute__((ext_vector_type(8)))  __bf16 v8bf;
typedef __attribute__((ext_vector_type(4)))  __bf16 v4bf;
typedef __attribute__((ext_vector_type(8)))  float  v8f;

#define BATCH 4096
#define DIN   2048
#define UNITS 2048
#define DCAT  4096   // UNITS + DIN
#define LDK   4096   // K dim of the fused GEMM
#define NK    (LDK / 32)

// LDS row stride (elements) for 32-K tiles: 32 + 8 pad -> 80B, conflict-free
#define LDS_STRIDE 40
#define SA_SZ (128 * LDS_STRIDE)          // one A stage buffer (elements)
#define SB_SZ (4 * 64 * LDS_STRIDE)       // one B stage buffer (elements)
#define SA_BYTES (SA_SZ * 2)
#define SB_BYTES (SB_SZ * 2)

__device__ __forceinline__ float fast_sigmoid(float x) {
  return 1.0f / (1.0f + __expf(-x));
}
__device__ __forceinline__ float fast_tanh(float x) {
  return 2.0f / (1.0f + __expf(-2.0f * x)) - 1.0f;
}

// Low 32 bits of a generic shared-aperture address = wave-relative LDS offset.
__device__ __forceinline__ uint32_t lds_off32(const void* p) {
  return (uint32_t)(uintptr_t)p;
}

// ---------------------------------------------------------------------------
// Prepass 1: A_bf16[b][k] = k < UNITS ? hidden[b][k] : x[b][k-UNITS]
// ---------------------------------------------------------------------------
__global__ __launch_bounds__(256) void k_build_a(
    const float* __restrict__ x, const float* __restrict__ h,
    __bf16* __restrict__ A) {
  size_t idx = ((size_t)blockIdx.x * 256 + threadIdx.x) * 4;
  int row = (int)(idx >> 12);      // / DCAT
  int col = (int)(idx & (DCAT - 1));
  const float* src = (col < UNITS)
      ? (h + (size_t)row * UNITS + col)
      : (x + (size_t)row * DIN + (col - UNITS));
  float4 v = *(const float4*)src;
  v4bf o;
  o[0] = (__bf16)v.x; o[1] = (__bf16)v.y; o[2] = (__bf16)v.z; o[3] = (__bf16)v.w;
  *(v4bf*)(A + idx) = o;
}

// ---------------------------------------------------------------------------
// Prepass 2: Wt_bf16[g*UNITS + u][k] = W_g[k][u]   (transpose + fp32->bf16)
// ---------------------------------------------------------------------------
__global__ __launch_bounds__(256) void k_build_wt(
    const float* __restrict__ Wf, const float* __restrict__ Wi,
    const float* __restrict__ Wo, const float* __restrict__ Wc,
    __bf16* __restrict__ Wt) {
  __shared__ float tile[32][33];
  int g = blockIdx.z;
  const float* W = (g == 0) ? Wf : (g == 1) ? Wi : (g == 2) ? Wo : Wc;
  int k0 = blockIdx.x * 32;   // over DCAT (gridDim.x = 128)
  int u0 = blockIdx.y * 32;   // over UNITS (gridDim.y = 64)
  int c  = threadIdx.x & 31;
  int rr = threadIdx.x >> 5;  // 0..7
#pragma unroll
  for (int i = 0; i < 4; i++) {
    int r = rr + i * 8;
    tile[r][c] = W[(size_t)(k0 + r) * UNITS + u0 + c];
  }
  __syncthreads();
#pragma unroll
  for (int i = 0; i < 4; i++) {
    int r = rr + i * 8;  // u offset
    Wt[((size_t)(g * UNITS + u0 + r)) * LDK + k0 + c] = (__bf16)tile[c][r];
  }
}

// ---------------------------------------------------------------------------
// Fused gate GEMM + LSTM epilogue.
// Block: 256 threads = 8 wave32 in a 4(m) x 2(n) grid.
// Block tile: 128(M) x 64(N per gate) x 4 gates; wave tile 32x32 per gate.
// Per 32-K step each wave issues 16 WMMAs from 2 A frags x 8 B frags.
// Triple-buffered async global->LDS staging; 1 barrier/K-step; prefetch
// distance of 2 K-steps (s_wait_asynccnt 6 keeps one stage in flight).
// ---------------------------------------------------------------------------
__global__ __launch_bounds__(256) void k_lstm_gemm(
    const __bf16* __restrict__ A,   // [BATCH][DCAT] bf16
    const __bf16* __restrict__ Wt,  // [4*UNITS][DCAT] bf16 (transposed W)
    const float* __restrict__ bfv, const float* __restrict__ biv,
    const float* __restrict__ bov, const float* __restrict__ bcv,
    const float* __restrict__ cell,
    float* __restrict__ out_h, float* __restrict__ out_c) {
  __shared__ __align__(16) __bf16 sA[3 * SA_SZ];
  __shared__ __align__(16) __bf16 sB[3 * SB_SZ];

  const int t    = threadIdx.x;
  const int lane = t & 31;
  const int wave = t >> 5;
  const int wm   = wave & 3;   // 4 m-waves -> 32 rows each
  const int wn   = wave >> 2;  // 2 n-waves -> 32 cols each

  const int rowA0 = blockIdx.y * 128;  // batch tile base
  const int colU0 = blockIdx.x * 64;   // unit tile base

  // A-fragment lane mapping: M = lane%16; K chunks at (lane/16)*8 and +16
  const int aM = lane & 15;
  const int aK = (lane >> 4) * 8;
  // B-fragment lane mapping: N = lane%16; K chunks at (lane/16)*8 and +16
  const int bN = lane & 15;
  const int bK = (lane >> 4) * 8;

  v8f acc[4][2][2];  // [gate][n-subtile][m-subtile]
#pragma unroll
  for (int g = 0; g < 4; g++)
#pragma unroll
    for (int ns = 0; ns < 2; ns++)
#pragma unroll
      for (int ms = 0; ms < 2; ms++) acc[g][ns][ms] = (v8f)0.0f;

  // ---- async staging state: 6 slots of 16B per thread per K-step ----
  // slots 0,1 : A tile (128x32 bf16);  slots 2..5 : B gate tiles (64x32 each)
  uint64_t g_src[6];
  uint32_t l_dst[6];  // byte offset within buffer 0
  {
    int i0 = t, i1 = t + 256;
    int r0 = i0 >> 2, c0 = (i0 & 3) * 8;
    int r1 = i1 >> 2, c1 = (i1 & 3) * 8;
    g_src[0] = (uint64_t)(uintptr_t)(A + (size_t)(rowA0 + r0) * LDK + c0);
    g_src[1] = (uint64_t)(uintptr_t)(A + (size_t)(rowA0 + r1) * LDK + c1);
    l_dst[0] = lds_off32(sA) + (uint32_t)(r0 * LDS_STRIDE + c0) * 2;
    l_dst[1] = lds_off32(sA) + (uint32_t)(r1 * LDS_STRIDE + c1) * 2;
    int br = t >> 2, bc_ = (t & 3) * 8;
#pragma unroll
    for (int g = 0; g < 4; g++) {
      g_src[2 + g] = (uint64_t)(uintptr_t)(
          Wt + ((size_t)(g * UNITS + colU0 + br)) * LDK + bc_);
      l_dst[2 + g] =
          lds_off32(sB) + (uint32_t)((g * 64 + br) * LDS_STRIDE + bc_) * 2;
    }
  }

  // Issue async copies for one K-step into buffer `buf`, advance src by 64B.
  auto stage = [&](int buf) {
#pragma unroll
    for (int i = 0; i < 6; i++) {
      uint32_t ldst =
          l_dst[i] + (uint32_t)buf * (i < 2 ? SA_BYTES : SB_BYTES);
      asm volatile("global_load_async_to_lds_b128 %0, %1, off"
                   :
                   : "v"(ldst), "v"(g_src[i])
                   : "memory");
      g_src[i] += 64;  // next 32-K chunk
    }
  };

  stage(0);
  stage(1);

  for (int it = 0; it < NK; ++it) {
    const int cur = it % 3;
    // Need buf[cur] landed; buf[cur+1]'s 6 async ops may stay in flight
    // (async loads complete in order).
    if (it < NK - 1)
      asm volatile("s_wait_asynccnt 0x6" ::: "memory");
    else
      asm volatile("s_wait_asynccnt 0x0" ::: "memory");
    __syncthreads();  // buf[cur] visible to all; buf[(it+2)%3] readers done
    if (it + 2 < NK) stage((it + 2) % 3);

    const __bf16* bufA = sA + cur * SA_SZ;
    const __bf16* bufB = sB + cur * SB_SZ;

    // two A fragments (shared across gates and n-subtiles)
    v16bf afrag[2];
#pragma unroll
    for (int ms = 0; ms < 2; ms++) {
      const __bf16* pA = bufA + (wm * 32 + ms * 16 + aM) * LDS_STRIDE + aK;
      v8bf alo = *(const v8bf*)pA;
      v8bf ahi = *(const v8bf*)(pA + 16);
#pragma unroll
      for (int i = 0; i < 8; i++) {
        afrag[ms][i] = alo[i];
        afrag[ms][i + 8] = ahi[i];
      }
    }

    // B fragments: 2-deep rotation so frag j+1's ds_loads can be in flight
    // while frag j's WMMAs execute.
    auto load_bfrag = [&](int j) -> v16bf {
      int g = j >> 1, ns = j & 1;
      const __bf16* pB =
          bufB + (g * 64 + wn * 32 + ns * 16 + bN) * LDS_STRIDE + bK;
      v8bf blo = *(const v8bf*)pB;
      v8bf bhi = *(const v8bf*)(pB + 16);
      v16bf r;
#pragma unroll
      for (int i = 0; i < 8; i++) { r[i] = blo[i]; r[i + 8] = bhi[i]; }
      return r;
    };

    v16bf bfr[2];
    bfr[0] = load_bfrag(0);
#pragma unroll
    for (int j = 0; j < 8; j++) {
      if (j < 7) bfr[(j + 1) & 1] = load_bfrag(j + 1);
      const int g = j >> 1, ns = j & 1;
#pragma unroll
      for (int ms = 0; ms < 2; ms++) {
        acc[g][ns][ms] = __builtin_amdgcn_wmma_f32_16x16x32_bf16(
            false, afrag[ms], false, bfr[j & 1], (short)0, acc[g][ns][ms],
            false, false);
      }
    }
  }

  // ---- Epilogue: gates + cell update, write next_hidden / next_cell ----
  // C/D layout: VGPR r -> M = r + (lane>=16 ? 8 : 0); N = lane%16
  const int mhalf = (lane >> 4) * 8;
#pragma unroll
  for (int ns = 0; ns < 2; ns++) {
    const int col = colU0 + wn * 32 + ns * 16 + (lane & 15);
    const float bF = bfv[col], bI = biv[col], bO = bov[col], bC = bcv[col];
#pragma unroll
    for (int ms = 0; ms < 2; ms++) {
      const int row0 = rowA0 + wm * 32 + ms * 16 + mhalf;
#pragma unroll
      for (int r = 0; r < 8; r++) {
        const int row = row0 + r;
        const float zf = acc[0][ns][ms][r] + bF;
        const float zi = acc[1][ns][ms][r] + bI;
        const float zo = acc[2][ns][ms][r] + bO;
        const float zc = acc[3][ns][ms][r] + bC;
        const float f  = fast_sigmoid(zf);
        const float ig = fast_sigmoid(zi);
        const float o  = fast_sigmoid(zo);
        const float cc = fast_tanh(zc);
        const float cprev = cell[(size_t)row * UNITS + col];
        const float cn = f * cprev + ig * cc;
        const float hn = o * fast_tanh(cn);
        out_c[(size_t)row * UNITS + col] = cn;
        out_h[(size_t)row * UNITS + col] = hn;
      }
    }
  }
}

// ---------------------------------------------------------------------------
extern "C" void kernel_launch(void* const* d_in, const int* in_sizes, int n_in,
                              void* d_out, int out_size, void* d_ws,
                              size_t ws_size, hipStream_t stream) {
  (void)in_sizes; (void)n_in; (void)out_size; (void)ws_size;
  const float* x    = (const float*)d_in[0];
  const float* h    = (const float*)d_in[1];
  const float* cell = (const float*)d_in[2];
  const float* Wf   = (const float*)d_in[3];
  const float* bf   = (const float*)d_in[4];
  const float* Wi   = (const float*)d_in[5];
  const float* bi   = (const float*)d_in[6];
  const float* Wo   = (const float*)d_in[7];
  const float* bo   = (const float*)d_in[8];
  const float* Wc   = (const float*)d_in[9];
  const float* bc   = (const float*)d_in[10];

  __bf16* Abf = (__bf16*)d_ws;                                     // 32 MB
  __bf16* Wt  = (__bf16*)((char*)d_ws + (size_t)BATCH * DCAT * 2); // 64 MB

  float* out_h = (float*)d_out;
  float* out_c = out_h + (size_t)BATCH * UNITS;

  // Prepass 1: concat + fp32->bf16 (16.7M elems, 4/thread)
  k_build_a<<<(BATCH * (size_t)DCAT) / (256 * 4), 256, 0, stream>>>(x, h, Abf);

  // Prepass 2: weight transpose + convert
  dim3 g2(DCAT / 32, UNITS / 32, 4);
  k_build_wt<<<g2, 256, 0, stream>>>(Wf, Wi, Wo, Wc, Wt);

  // Fused GEMM + LSTM epilogue
  dim3 g3(UNITS / 64, BATCH / 128);
  k_lstm_gemm<<<g3, 256, 0, stream>>>(Abf, Wt, bf, bi, bo, bc, cell, out_h,
                                      out_c);
}
